// CRPExpertAggregator_77945066488255
// MI455X (gfx1250) — matrix operand order repaired and checked by reading, not compiled, not run on hardware
//
#include <hip/hip_runtime.h>
#include <hip/hip_bf16.h>

typedef __attribute__((ext_vector_type(16))) _Float16 v16h;
typedef __attribute__((ext_vector_type(8)))  _Float16 v8h;
typedef __attribute__((ext_vector_type(8)))  float    v8f;

#define NUM_SLOTS   64
#define AGENT_DIM   256
#define EMBED_DIM   256
#define N_QUERIES   4
#define N_EXPERTS   16
#define NUM_CLASSES 100
#define BATCH       256

#define HS_STRIDE 264            // f16 elems per H row (256 + 8 pad, keeps 16B align)
#define KT_STRIDE 257            // f16 elems per K/V row (256 + 1 pad, conflict-free)
#define HS_ELEMS (NUM_SLOTS * HS_STRIDE)     // 16896 halves
#define KT_ELEMS (NUM_SLOTS * KT_STRIDE)     // 16448 halves
#define SMEM_BYTES ((HS_ELEMS + 2 * KT_ELEMS) * 2 + (4 * 64 + 4 * 256) * 4) // 104704 B

// ---------------------------------------------------------------------------
// Kernel 0: transpose+downconvert Wk/Wv (E,d,A) f32 -> (E,A,d) f16 so that
// WMMA B-fragment loads are coalesced across lanes (lane = d, k stride 256).
// ---------------------------------------------------------------------------
__global__ void prep_weights(const float* __restrict__ Wk, const float* __restrict__ Wv,
                             _Float16* __restrict__ WTk, _Float16* __restrict__ WTv) {
    int idx = blockIdx.x * 256 + threadIdx.x;      // = e*65536 + a*256 + d
    int e = idx >> 16;
    int a = (idx >> 8) & 255;
    int d = idx & 255;
    int src = (e << 16) + (d << 8) + a;
    WTk[idx] = (_Float16)Wk[src];
    WTv[idx] = (_Float16)Wv[src];
}

// ---------------------------------------------------------------------------
// Kernel 1: one workgroup per (b, e). Fused: K/V projection (WMMA f16->f32),
// scores, softmax, attn@V, mean over queries -> z[b,e,:].
// K and V GEMMs share A-fragments: load each A-frag from LDS once per ka-step
// and feed both the Wk and Wv B-streams (halves DS traffic per WMMA).
// ---------------------------------------------------------------------------
__global__ void __launch_bounds__(256) fused_expert_attn(
    const float* __restrict__ x,
    const float* __restrict__ queries,
    const _Float16* __restrict__ WTk,
    const _Float16* __restrict__ WTv,
    float* __restrict__ z)
{
    extern __shared__ __align__(16) char smem_raw[];
    _Float16* Hs   = (_Float16*)smem_raw;          // [64][264] f16  (A operand)
    _Float16* Kt   = Hs + HS_ELEMS;                // [64][257] f16  (K tile)
    _Float16* Vt   = Kt + KT_ELEMS;                // [64][257] f16  (V tile)
    float*    attn = (float*)(Vt + KT_ELEMS);      // [4][64] scores -> softmax
    float*    qs   = attn + 4 * 64;                // [4][256] staged queries

    const int t   = threadIdx.x;
    const int blk = blockIdx.x;                    // blk = b*16 + e
    const int b   = blk >> 4;
    const int e   = blk & 15;

    // ---- stage H[b] (64x256 f32 -> f16, padded rows) and queries[e] ----
    const float* xb = x + (size_t)b * (NUM_SLOTS * AGENT_DIM);
    for (int idx = t; idx < NUM_SLOTS * AGENT_DIM; idx += 256) {
        int s = idx >> 8, a = idx & 255;
        Hs[s * HS_STRIDE + a] = (_Float16)xb[idx];
    }
    const float* qe = queries + (size_t)e * (N_QUERIES * EMBED_DIM);
    for (int idx = t; idx < N_QUERIES * EMBED_DIM; idx += 256) qs[idx] = qe[idx];
    __syncthreads();

    // ---- K/V = H @ W^T via v_wmma_f32_16x16x32_f16 (fused K+V passes) ----
    const int w    = t >> 5;
    const int lane = t & 31;
    const int mrow = lane & 15;           // M row within 16x16 tile
    const int hi8  = (lane >> 4) << 3;    // A-frag K offset / C-frag M offset

    const _Float16* WKe = WTk + (size_t)e * (AGENT_DIM * EMBED_DIM);
    const _Float16* WVe = WTv + (size_t)e * (AGENT_DIM * EMBED_DIM);

    for (int half = 0; half < 2; ++half) {
        const int nd = w + half * 8;      // this wave's N-block (of 16)
        v8f acck[4] = {};
        v8f accv[4] = {};
        for (int ka = 0; ka < 8; ++ka) {
            // B fragments for both streams: lane = N column, K in table order (8h+s, 16+8h+s)
            const size_t boff = (size_t)(ka * 32 + hi8) * EMBED_DIM + nd * 16 + mrow;
            v16h bk, bv;
            #pragma unroll
            for (int i = 0; i < 8; ++i) {
                bk[i] = WKe[boff + i * EMBED_DIM]; bk[i + 8] = WKe[boff + (i + 16) * EMBED_DIM];
                bv[i] = WVe[boff + i * EMBED_DIM]; bv[i + 8] = WVe[boff + (i + 16) * EMBED_DIM];
            }
            // A fragments: loaded ONCE, used by both K and V WMMAs
            const _Float16* hp = Hs + mrow * HS_STRIDE + ka * 32 + hi8;
            v16h af[4];
            #pragma unroll
            for (int mi = 0; mi < 4; ++mi) {
                const _Float16* ap = hp + mi * 16 * HS_STRIDE;
                v8h lo = *(const v8h*)(ap);
                v8h hi = *(const v8h*)(ap + 16);
                #pragma unroll
                for (int i = 0; i < 8; ++i) { af[mi][i] = lo[i]; af[mi][i + 8] = hi[i]; }
            }
            #pragma unroll
            for (int mi = 0; mi < 4; ++mi)
                acck[mi] = __builtin_amdgcn_wmma_f32_16x16x32_f16(
                    false, af[mi], false, bk, (short)0, acck[mi], false, false);
            #pragma unroll
            for (int mi = 0; mi < 4; ++mi)
                accv[mi] = __builtin_amdgcn_wmma_f32_16x16x32_f16(
                    false, af[mi], false, bv, (short)0, accv[mi], false, false);
        }
        // C fragment scatter: VGPR r, lane l -> (m = r + 8*(l>=16), n = l&15)
        #pragma unroll
        for (int mi = 0; mi < 4; ++mi) {
            #pragma unroll
            for (int r = 0; r < 8; ++r) {
                int m = mi * 16 + r + hi8;
                int n = nd * 16 + mrow;
                Kt[m * KT_STRIDE + n] = (_Float16)acck[mi][r];
                Vt[m * KT_STRIDE + n] = (_Float16)accv[mi][r];
            }
        }
    }
    __syncthreads();

    // ---- scores[l][s] = (q_l . K_s) / sqrt(256) ----
    {
        int l = t >> 6, s = t & 63;
        const float* q = qs + l * EMBED_DIM;
        const _Float16* kr = Kt + s * KT_STRIDE;
        float sc = 0.f;
        for (int d = 0; d < EMBED_DIM; ++d) sc = fmaf(q[d], (float)kr[d], sc);
        attn[l * 64 + s] = sc * 0.0625f;
    }
    __syncthreads();
    // ---- softmax over s (one thread per query row; 64 elems, trivial) ----
    if (t < N_QUERIES) {
        float* row = attn + t * 64;
        float m = row[0];
        for (int s = 1; s < 64; ++s) m = fmaxf(m, row[s]);
        float sum = 0.f;
        for (int s = 0; s < 64; ++s) { float ex = __expf(row[s] - m); row[s] = ex; sum += ex; }
        float inv = 1.f / sum;
        for (int s = 0; s < 64; ++s) row[s] *= inv;
    }
    __syncthreads();

    // ---- z[b,e,d] = mean_l sum_s attn[l,s] * V[s,d]  (thread t owns d=t) ----
    {
        int d = t;
        float a0 = 0.f, a1 = 0.f, a2 = 0.f, a3 = 0.f;
        for (int s = 0; s < 64; ++s) {
            float v = (float)Vt[s * KT_STRIDE + d];
            a0 = fmaf(attn[s],       v, a0);
            a1 = fmaf(attn[64 + s],  v, a1);
            a2 = fmaf(attn[128 + s], v, a2);
            a3 = fmaf(attn[192 + s], v, a3);
        }
        z[(size_t)blk * EMBED_DIM + d] = 0.25f * ((a0 + a1) + (a2 + a3));
    }
}

// ---------------------------------------------------------------------------
// Kernel 2: per-batch gating + logits. raw=||z||, score=raw*log1p(cnt+1),
// top-3 softmax gate, final = sum gate*z, logits = final @ cq^T.
// ---------------------------------------------------------------------------
__global__ void __launch_bounds__(256) epilogue(
    const float* __restrict__ z,
    const float* __restrict__ cq,
    const int* __restrict__ counts,
    float* __restrict__ out)
{
    __shared__ float red[256];
    __shared__ float allsc[N_EXPERTS];
    __shared__ float gate[3];
    __shared__ int   sidx[3];
    __shared__ float fin[EMBED_DIM];

    const int t = threadIdx.x;
    const int b = blockIdx.x;
    const float* zb = z + (size_t)b * (N_EXPERTS * EMBED_DIM);

    { // partial sum of squares: 16 chunks per expert
        int e = t >> 4, c = t & 15;
        const float* p = zb + e * EMBED_DIM + c * 16;
        float s = 0.f;
        for (int j = 0; j < 16; ++j) s = fmaf(p[j], p[j], s);
        red[t] = s;
    }
    __syncthreads();
    if (t < N_EXPERTS) {
        float s = 0.f;
        for (int c = 0; c < 16; ++c) s += red[t * 16 + c];
        allsc[t] = sqrtf(s) * logf((float)counts[t] + 2.0f);   // log1p(cnt+1)
    }
    __syncthreads();
    if (t == 0) {  // top-3 of 16 (ties -> lower index, matches lax.top_k)
        float s0 = -1e30f, s1 = -1e30f, s2 = -1e30f; int i0 = 0, i1 = 0, i2 = 0;
        for (int ee = 0; ee < N_EXPERTS; ++ee) {
            float v = allsc[ee];
            if (v > s0)      { s2 = s1; i2 = i1; s1 = s0; i1 = i0; s0 = v; i0 = ee; }
            else if (v > s1) { s2 = s1; i2 = i1; s1 = v; i1 = ee; }
            else if (v > s2) { s2 = v; i2 = ee; }
        }
        float e0 = 1.0f, e1 = __expf(s1 - s0), e2 = __expf(s2 - s0);
        float inv = 1.f / (e0 + e1 + e2);
        gate[0] = e0 * inv; gate[1] = e1 * inv; gate[2] = e2 * inv;
        sidx[0] = i0; sidx[1] = i1; sidx[2] = i2;
    }
    __syncthreads();
    fin[t] = gate[0] * zb[sidx[0] * EMBED_DIM + t]
           + gate[1] * zb[sidx[1] * EMBED_DIM + t]
           + gate[2] * zb[sidx[2] * EMBED_DIM + t];
    __syncthreads();
    if (t < NUM_CLASSES) {
        const float* cr = cq + t * EMBED_DIM;
        float acc = 0.f;
        for (int d = 0; d < EMBED_DIM; ++d) acc = fmaf(fin[d], cr[d], acc);
        out[(size_t)b * NUM_CLASSES + t] = acc;
    }
}

// ---------------------------------------------------------------------------
extern "C" void kernel_launch(void* const* d_in, const int* in_sizes, int n_in,
                              void* d_out, int out_size, void* d_ws, size_t ws_size,
                              hipStream_t stream) {
    const float* x      = (const float*)d_in[0];
    const float* q      = (const float*)d_in[1];
    const float* Wk     = (const float*)d_in[2];
    const float* Wv     = (const float*)d_in[3];
    const float* cq     = (const float*)d_in[4];
    const int*   counts = (const int*)d_in[5];
    float* out = (float*)d_out;

    // workspace: WTk (2MB f16) | WTv (2MB f16) | z (4MB f32) = 8MB
    const size_t wsz = (size_t)N_EXPERTS * AGENT_DIM * EMBED_DIM;
    _Float16* WTk = (_Float16*)d_ws;
    _Float16* WTv = WTk + wsz;
    float*    z   = (float*)((char*)d_ws + 2 * wsz * sizeof(_Float16));

    prep_weights<<<dim3((unsigned)(wsz / 256)), dim3(256), 0, stream>>>(Wk, Wv, WTk, WTv);
    fused_expert_attn<<<dim3(BATCH * N_EXPERTS), dim3(256), SMEM_BYTES, stream>>>(x, q, WTk, WTv, z);
    epilogue<<<dim3(BATCH), dim3(256), 0, stream>>>(z, cq, counts, out);
}
